// TimeConv_48644799594893
// MI455X (gfx1250) — compile-verified, matrix operand order checked by reading, not run on hardware
//
#include <hip/hip_runtime.h>
#include <hip/hip_bf16.h>

// ---------------------------------------------------------------------------
// MI455X / gfx1250 implementation.
//  - persistent hidden state h kept in FP16 in d_ws (128 MB, L2-resident)
//  - all matmuls via v_wmma_f32_16x16x32_f16 (f32 accumulate)
//  - gather-mean uses packed f16 VALU (v_pk_add_f16) to cut VALU pressure
//  - weights pre-transposed + fused once per launch by a prep kernel
//  - one kernel per topological level (levels are serially dependent)
// ---------------------------------------------------------------------------

#define NTOT 500000
#define LVLS 10
#define NPL  50000
#define KPRE 8
#define INF  32
#define HID  128

typedef _Float16 v16h __attribute__((ext_vector_type(16)));
typedef float    v8f  __attribute__((ext_vector_type(8)));
typedef _Float16 v2h  __attribute__((ext_vector_type(2)));

union U32H2 { unsigned u; v2h h; };
union F16x16 { v16h v; unsigned u[8]; };

__device__ __forceinline__ float leaky(float v) { return v > 0.f ? v : 0.1f * v; }

__device__ __forceinline__ unsigned pack2h(float a, float b) {
    U32H2 t; t.h[0] = (_Float16)a; t.h[1] = (_Float16)b; return t.u;
}

// K element index for fragment slot j (0..7 u32 pairs), lane-half, K-base k0.
// Matches ISA 7.12.2 16-bit A 16x32 layout; B assumed symmetric (N across lanes).
__device__ __forceinline__ int frag_kk(int j, int half, int k0) {
    return k0 + ((j >> 2) << 4) + (half << 3) + ((j & 3) << 1);
}

__device__ __forceinline__ v16h load_frag(const _Float16* row, int k0, int half) {
    F16x16 f;
#pragma unroll
    for (int j = 0; j < 8; ++j)
        f.u[j] = *(const unsigned*)(row + frag_kk(j, half, k0));
    return f.v;
}

__device__ __forceinline__ v8f wmma_f16(v16h a, v16h b, v8f c) {
    return __builtin_amdgcn_wmma_f32_16x16x32_f16(false, a, false, b, (short)0, c,
                                                  false, false);
}

// ---------------------------------------------------------------------------
// Workspace layout (bytes):
//   [0, 128000000)            h16: NTOT x HID  f16
//   [128000000, +118784)      wf16: packed f16 weights (all [N][K] transposed)
//       W1n_t   64x128  @ half-off 0
//       W1s_t   64x32   @ 8192
//       W2cat_t 128x128 @ 10240   (rows k<64 = neigh W2, k>=64 = self W2)
//       Wpi2_t  128x64  @ 26624
//       Wout1_t 128x192 @ 34816   (k<128 = out W1 top; k>=128 = W2g @ out W1 bot)
//   then bf32 region (f32):
//       b1n@0 b1s@64 b2cat@128 b2pi@256 w1pi@384 b1pi@448
//       w1g@512 b1g@576 bout1@640 w2out@768 b2out@896
// ---------------------------------------------------------------------------
#define WF16_OFF   128000000
#define WF16_BYTES 118784
#define NGROUP     (NPL / 16)  // 3125

struct PP { const float* a[20]; };
// a[]: piW1 piB1 piW2 piB2 | nW1 nB1 nW2 nB2 | sW1 sB1 sW2 sB2 |
//      gW1 gB1 gW2 gB2 | oW1 oB1 oW2 oB2

__global__ void prep_kernel(PP pp, _Float16* __restrict__ wf16,
                            float* __restrict__ bf32) {
    const int tid = threadIdx.x, nt = blockDim.x;
    const float* nW1 = pp.a[4];  const float* sW1 = pp.a[8];
    const float* nW2 = pp.a[6];  const float* sW2 = pp.a[10];
    const float* piW2 = pp.a[2]; const float* gW2 = pp.a[14];
    const float* oW1 = pp.a[16];

    for (int i = tid; i < 64 * 128; i += nt) {            // W1n_t [n=64][k=128]
        int n = i / 128, k = i % 128;
        wf16[i] = (_Float16)nW1[k * 64 + n];
    }
    for (int i = tid; i < 64 * 32; i += nt) {             // W1s_t [n=64][k=32]
        int n = i / 32, k = i % 32;
        wf16[8192 + i] = (_Float16)sW1[k * 64 + n];
    }
    for (int i = tid; i < 128 * 128; i += nt) {           // W2cat_t [n=128][k=128]
        int n = i / 128, k = i % 128;
        float v = (k < 64) ? nW2[k * 128 + n] : sW2[(k - 64) * 128 + n];
        wf16[10240 + i] = (_Float16)v;
    }
    for (int i = tid; i < 128 * 64; i += nt) {            // Wpi2_t [n=128][k=64]
        int n = i / 64, k = i % 64;
        wf16[26624 + i] = (_Float16)piW2[k * 128 + n];
    }
    for (int i = tid; i < 128 * 192; i += nt) {           // Wout1_t [n=128][k=192]
        int n = i / 192, k = i % 192;
        float v;
        if (k < 128) {
            v = oW1[k * 128 + n];
        } else {
            int kk = k - 128; float s = 0.f;
            for (int t = 0; t < 128; ++t) s += gW2[kk * 128 + t] * oW1[(128 + t) * 128 + n];
            v = s;
        }
        wf16[34816 + i] = (_Float16)v;
    }
    // biases / small vectors
    for (int i = tid; i < 64; i += nt)  bf32[i]       = pp.a[5][i];            // b1n
    for (int i = tid; i < 64; i += nt)  bf32[64 + i]  = pp.a[9][i];            // b1s
    for (int i = tid; i < 128; i += nt) bf32[128 + i] = pp.a[7][i] + pp.a[11][i]; // b2cat
    for (int i = tid; i < 128; i += nt) bf32[256 + i] = pp.a[3][i];            // b2pi
    for (int i = tid; i < 64; i += nt)  bf32[384 + i] = pp.a[0][i];            // w1pi
    for (int i = tid; i < 64; i += nt)  bf32[448 + i] = pp.a[1][i];            // b1pi
    for (int i = tid; i < 64; i += nt)  bf32[512 + i] = pp.a[12][i];           // w1g
    for (int i = tid; i < 64; i += nt)  bf32[576 + i] = pp.a[13][i];           // b1g
    for (int i = tid; i < 128; i += nt) {                                      // bout1
        float s = pp.a[17][i];
        for (int k = 0; k < 128; ++k) s += pp.a[15][k] * oW1[(128 + k) * 128 + i];
        bf32[640 + i] = s;
    }
    for (int i = tid; i < 128; i += nt) bf32[768 + i] = pp.a[18][i];           // w2out
    if (tid == 0) bf32[896] = pp.a[19][0];                                     // b2out
}

// ---------------------------------------------------------------------------
// Level 0: h = mlp2(delay, pi).  din=1 -> layer1 is VALU, layer2 via WMMA.
// ---------------------------------------------------------------------------
__global__ void level0_kernel(_Float16* __restrict__ h16,
                              const float* __restrict__ delay,
                              const _Float16* __restrict__ wf16,
                              const float* __restrict__ bf32) {
    __shared__ _Float16 sW[128 * 64];     // Wpi2_t
    __shared__ _Float16 sx[4][16 * 64];
    {
        const unsigned* src = (const unsigned*)(wf16 + 26624);
        unsigned* dst = (unsigned*)sW;
        for (int i = threadIdx.x; i < (128 * 64) / 2; i += blockDim.x) dst[i] = src[i];
    }
    __syncthreads();

    const int wave = threadIdx.x >> 5, lane = threadIdx.x & 31;
    const int group = blockIdx.x * 4 + wave;
    if (group >= NGROUP) return;

    const int half = lane >> 4, nl = lane & 15;
    const float* w1 = bf32 + 384;
    const float* b1 = bf32 + 448;
    const float* b2 = bf32 + 256;
    _Float16* x0 = sx[wave];
    const int node0 = group * 16;

    for (int m = 0; m < 16; ++m) {
        float d = delay[node0 + m];
        int j = lane * 2;
        float v0 = leaky(d * w1[j] + b1[j]);
        float v1 = leaky(d * w1[j + 1] + b1[j + 1]);
        *(unsigned*)(x0 + m * 64 + j) = pack2h(v0, v1);
    }

#pragma unroll
    for (int ntile = 0; ntile < 8; ++ntile) {
        v8f c = {};
        const _Float16* brow = sW + (ntile * 16 + nl) * 64;
#pragma unroll
        for (int ks = 0; ks < 2; ++ks) {
            v16h a = load_frag(x0 + nl * 64, ks * 32, half);
            v16h b = load_frag(brow, ks * 32, half);
            c = wmma_f16(a, b, c);
        }
        float bias = b2[ntile * 16 + nl];
#pragma unroll
        for (int r = 0; r < 8; ++r) {
            long row = node0 + r + 8 * half;
            h16[row * HID + ntile * 16 + nl] = (_Float16)(c[r] + bias);
        }
    }
}

// ---------------------------------------------------------------------------
// Levels 1..9: gather-mean + (128->64->128) + (32->64->128) fused layer2.
// ---------------------------------------------------------------------------
__global__ void level_kernel(int lvl, _Float16* __restrict__ h16,
                             const float* __restrict__ feat,
                             const int* __restrict__ src_idx,
                             const _Float16* __restrict__ wf16,
                             const float* __restrict__ bf32, int relu_flag) {
    extern __shared__ char dynsm[];
    _Float16* smemh = (_Float16*)dynsm;
    _Float16* sW1n = smemh;            // 64*128
    _Float16* sW1s = smemh + 8192;     // 64*32
    _Float16* sW2  = smemh + 10240;    // 128*128
    const int WSH = 26624;
    {
        const unsigned* src = (const unsigned*)wf16;
        unsigned* dst = (unsigned*)smemh;
        for (int i = threadIdx.x; i < WSH / 2; i += blockDim.x) dst[i] = src[i];
    }
    __syncthreads();

    const int wave = threadIdx.x >> 5, lane = threadIdx.x & 31;
    const int group = blockIdx.x * 4 + wave;
    if (group >= NGROUP) return;

    _Float16* xcat = smemh + WSH + wave * 4608;  // [16][160] f16
    _Float16* y1   = xcat + 2560;                // [16][128] f16
    const int half = lane >> 4, nl = lane & 15;
    const int node0 = lvl * NPL + group * 16;

    // ---- gather + mean of K=8 predecessor rows ----------------------------
    // packed-f16 accumulation: 2x v_pk_add_f16 per 8B instead of 8 f32 ops
    const int srcbase = (lvl - 1) * NPL * KPRE + group * 16 * KPRE;
    const _Float16 inv8 = (_Float16)0.125f;
    const v2h vscale = {inv8, inv8};
    for (int m = 0; m < 16; ++m) {
        v2h s01 = {(_Float16)0.f, (_Float16)0.f};
        v2h s23 = {(_Float16)0.f, (_Float16)0.f};
        const int* sp = src_idx + srcbase + m * KPRE;
#pragma unroll
        for (int p = 0; p < KPRE; ++p) {
            long pred = sp[p];
            uint2 d = *(const uint2*)(h16 + pred * HID + lane * 4);
            U32H2 lo, hi; lo.u = d.x; hi.u = d.y;
            s01 += lo.h;
            s23 += hi.h;
        }
        U32H2 r0, r1;
        r0.h = s01 * vscale;
        r1.h = s23 * vscale;
        uint2 r; r.x = r0.u; r.y = r1.u;
        *(uint2*)(xcat + m * 160 + lane * 4) = r;
    }
    // ---- self features (32 f32 -> f16) ------------------------------------
    for (int m = 0; m < 16; ++m)
        xcat[m * 160 + 128 + lane] = (_Float16)feat[(long)(node0 + m) * INF + lane];

    const float* b1n = bf32;
    const float* b1s = bf32 + 64;
    const float* b2c = bf32 + 128;

    // ---- layer 1, neighbor branch: 16x64 <- 16x128 @ 128x64 ---------------
#pragma unroll
    for (int ntile = 0; ntile < 4; ++ntile) {
        v8f c = {};
        const _Float16* brow = sW1n + (ntile * 16 + nl) * 128;
#pragma unroll
        for (int ks = 0; ks < 4; ++ks) {
            v16h a = load_frag(xcat + nl * 160, ks * 32, half);
            v16h b = load_frag(brow, ks * 32, half);
            c = wmma_f16(a, b, c);
        }
        float bias = b1n[ntile * 16 + nl];
#pragma unroll
        for (int r = 0; r < 8; ++r)
            y1[(r + 8 * half) * 128 + ntile * 16 + nl] = (_Float16)leaky(c[r] + bias);
    }
    // ---- layer 1, self branch: 16x64 <- 16x32 @ 32x64 ---------------------
#pragma unroll
    for (int ntile = 0; ntile < 4; ++ntile) {
        const _Float16* brow = sW1s + (ntile * 16 + nl) * 32;
        v16h a = load_frag(xcat + nl * 160, 128, half);
        v16h b = load_frag(brow, 0, half);
        v8f c = {};
        c = wmma_f16(a, b, c);
        float bias = b1s[ntile * 16 + nl];
#pragma unroll
        for (int r = 0; r < 8; ++r)
            y1[(r + 8 * half) * 128 + 64 + ntile * 16 + nl] =
                (_Float16)leaky(c[r] + bias);
    }
    // ---- layer 2 fused: 16x128 <- 16x128 @ 128x128 ------------------------
#pragma unroll
    for (int ntile = 0; ntile < 8; ++ntile) {
        v8f c = {};
        const _Float16* brow = sW2 + (ntile * 16 + nl) * 128;
#pragma unroll
        for (int ks = 0; ks < 4; ++ks) {
            v16h a = load_frag(y1 + nl * 128, ks * 32, half);
            v16h b = load_frag(brow, ks * 32, half);
            c = wmma_f16(a, b, c);
        }
        float bias = b2c[ntile * 16 + nl];
#pragma unroll
        for (int r = 0; r < 8; ++r) {
            float v = c[r] + bias;
            if (relu_flag) v = v > 0.f ? v : 0.f;
            long row = node0 + r + 8 * half;
            h16[row * HID + ntile * 16 + nl] = (_Float16)v;
        }
    }
}

// ---------------------------------------------------------------------------
// Output: y = mlp2(concat(h_po, mlp2(po_feat, glob)), out).
// Global branch folded into a single 16x128 <- 16x192 @ 192x128 GEMM.
// ---------------------------------------------------------------------------
__global__ void out_kernel(const _Float16* __restrict__ h16,
                           const int* __restrict__ po_idx,
                           const float* __restrict__ po_feat,
                           const _Float16* __restrict__ wf16,
                           const float* __restrict__ bf32,
                           float* __restrict__ out) {
    extern __shared__ char dynsm[];
    _Float16* sWo = (_Float16*)dynsm;                       // 128*192 halves
    {
        const unsigned* src = (const unsigned*)(wf16 + 34816);
        unsigned* dst = (unsigned*)sWo;
        for (int i = threadIdx.x; i < (128 * 192) / 2; i += blockDim.x) dst[i] = src[i];
    }
    __syncthreads();

    const int wave = threadIdx.x >> 5, lane = threadIdx.x & 31;
    const int group = blockIdx.x * 4 + wave;
    if (group >= NGROUP) return;

    _Float16* xo = (_Float16*)(dynsm + 49152) + wave * (16 * 192);
    float* y1o = (float*)(dynsm + 73728) + wave * (16 * 128);
    const int half = lane >> 4, nl = lane & 15;

    const float* w1g = bf32 + 512;
    const float* b1g = bf32 + 576;
    const float* bo1 = bf32 + 640;
    const float* w2o = bf32 + 768;
    const float b2o = bf32[896];

    for (int m = 0; m < 16; ++m) {
        int nodei = group * 16 + m;
        long po = po_idx[nodei];
        // h_gnn: raw f16 copy
        *(uint2*)(xo + m * 192 + lane * 4) = *(const uint2*)(h16 + po * HID + lane * 4);
        // global-branch hidden (64) under leaky-relu
        float pf = po_feat[nodei];
        int j = lane * 2;
        float v0 = leaky(pf * w1g[j] + b1g[j]);
        float v1 = leaky(pf * w1g[j + 1] + b1g[j + 1]);
        *(unsigned*)(xo + m * 192 + 128 + j) = pack2h(v0, v1);
    }

#pragma unroll
    for (int ntile = 0; ntile < 8; ++ntile) {
        v8f c = {};
        const _Float16* brow = sWo + (ntile * 16 + nl) * 192;
#pragma unroll
        for (int ks = 0; ks < 6; ++ks) {
            v16h a = load_frag(xo + nl * 192, ks * 32, half);
            v16h b = load_frag(brow, ks * 32, half);
            c = wmma_f16(a, b, c);
        }
        float bias = bo1[ntile * 16 + nl];
#pragma unroll
        for (int r = 0; r < 8; ++r)
            y1o[(r + 8 * half) * 128 + ntile * 16 + nl] = leaky(c[r] + bias);
    }

    // final 128 -> 1 dot per node, wave32 reduction
    for (int m = 0; m < 16; ++m) {
        float p = 0.f;
#pragma unroll
        for (int t = 0; t < 4; ++t)
            p += y1o[m * 128 + lane * 4 + t] * w2o[lane * 4 + t];
        for (int off = 16; off > 0; off >>= 1) p += __shfl_xor(p, off, 32);
        if (lane == 0) out[group * 16 + m] = p + b2o;
    }
}

// ---------------------------------------------------------------------------
extern "C" void kernel_launch(void* const* d_in, const int* in_sizes, int n_in,
                              void* d_out, int out_size, void* d_ws, size_t ws_size,
                              hipStream_t stream) {
    (void)in_sizes; (void)n_in; (void)out_size; (void)ws_size;
    const float* feat    = (const float*)d_in[0];
    const float* delay   = (const float*)d_in[1];
    /* d_in[2] topo: identity arange, addressing done arithmetically */
    const int*   src_idx = (const int*)d_in[3];
    /* d_in[4] is_po: PO == last level, encoded via relu_flag */
    const int*   po_idx  = (const int*)d_in[5];
    const float* po_feat = (const float*)d_in[6];

    PP pp;
    for (int i = 0; i < 20; ++i) pp.a[i] = (const float*)d_in[7 + i];

    char* ws = (char*)d_ws;
    _Float16* h16  = (_Float16*)ws;
    _Float16* wf16 = (_Float16*)(ws + WF16_OFF);
    float*    bf32 = (float*)(ws + WF16_OFF + WF16_BYTES);
    float*    out  = (float*)d_out;

    prep_kernel<<<1, 256, 0, stream>>>(pp, wf16, bf32);

    const int nb = (NGROUP + 3) / 4;  // 4 waves (tiles) per 128-thread block
    level0_kernel<<<nb, 128, 0, stream>>>(h16, delay, wf16, bf32);
    for (int lvl = 1; lvl < LVLS; ++lvl)
        level_kernel<<<nb, 128, 90112, stream>>>(lvl, h16, feat, src_idx, wf16, bf32,
                                                 lvl != LVLS - 1);
    out_kernel<<<nb, 128, 106496, stream>>>(h16, po_idx, po_feat, wf16, bf32, out);
}
